// ScaledDotProductAttention_80659485819079
// MI455X (gfx1250) — matrix operand order repaired and checked by reading, not compiled, not run on hardware
//
#include <hip/hip_runtime.h>
#include <math.h>

typedef _Float16 v4h  __attribute__((ext_vector_type(4)));
typedef _Float16 v8h  __attribute__((ext_vector_type(8)));
typedef _Float16 v16h __attribute__((ext_vector_type(16)));
typedef float    v8f  __attribute__((ext_vector_type(8)));

#define SEQ    1024
#define HDIM   64
#define QTILE  128                 // 8 waves x 16 queries
#define KVTILE 64
#define NKV    (SEQ / KVTILE)
#define KSTR   72                  // padded f16 stride (bank-conflict free frag loads)

// lane <-> lane^16 exchange, immediate-form ds_swizzle (xor=16, and=0x1f)
__device__ __forceinline__ float swz16f(float x) {
    return __builtin_bit_cast(float,
        __builtin_amdgcn_ds_swizzle(__builtin_bit_cast(int, x), 0x401f));
}
__device__ __forceinline__ v16h cat8(v8h a, v8h b) {
    return __builtin_shufflevector(a, b, 0,1,2,3,4,5,6,7,8,9,10,11,12,13,14,15);
}

__global__ __launch_bounds__(256)
void ScaledDotProductAttention_80659485819079_kernel(
        const float* __restrict__ Q, const float* __restrict__ Kp,
        const float* __restrict__ Vp, float* __restrict__ O) {
    __shared__ _Float16 kT[2][KVTILE * KSTR];   // [buf][key][d]  row-major keys
    __shared__ _Float16 vT[2][HDIM   * KSTR];   // [buf][d][key]  transposed values

    const int bh    = blockIdx.x;               // fused (b,h)
    const int qbase = blockIdx.y * QTILE;
    const int tid   = threadIdx.x;
    const int wave  = tid >> 5;
    const int lane  = tid & 31;
    const int half  = lane >> 4;
    const int l16   = lane & 15;
    const float scale = 0.25f;                  // 1/sqrt(num_heads=16)

    // ---- Q^T B-fragments, resident for whole kernel (scale folded in).
    const float* qrow = Q + ((size_t)bh * SEQ + qbase + wave * 16 + l16) * HDIM;
    v16h qb[2];
#pragma unroll
    for (int kf = 0; kf < 2; ++kf) {
        const int d0 = kf * 32 + half * 16;
#pragma unroll
        for (int i = 0; i < 16; ++i)
            qb[kf][i] = (_Float16)(qrow[d0 + i] * scale);
    }

    float mrun = -1e30f, lrun = 0.0f;           // per-lane (per-query) running stats
    v8f oacc[4];                                // O^T fragments: M=d, N=query
#pragma unroll
    for (int t = 0; t < 4; ++t)
#pragma unroll
        for (int v = 0; v < 8; ++v) oacc[t][v] = 0.0f;

    const float* kb = Kp + (size_t)bh * SEQ * HDIM;
    const float* vb = Vp + (size_t)bh * SEQ * HDIM;

    const int r0 = (tid >> 4) * 4;              // staging 4x4 block / thread
    const int c0 = (tid & 15) * 4;

    float kx[4][4], vx[4][4];                   // in-flight next tile (registers)

    // ---- prologue: load + stage tile 0 into buffer 0
#pragma unroll
    for (int i = 0; i < 4; ++i) {
        const float4 a = *(const float4*)(kb + (size_t)(r0 + i) * HDIM + c0);
        const float4 b = *(const float4*)(vb + (size_t)(r0 + i) * HDIM + c0);
        kx[i][0] = a.x; kx[i][1] = a.y; kx[i][2] = a.z; kx[i][3] = a.w;
        vx[i][0] = b.x; vx[i][1] = b.y; vx[i][2] = b.z; vx[i][3] = b.w;
    }
#pragma unroll
    for (int i = 0; i < 4; ++i) {
        v4h p;
#pragma unroll
        for (int j = 0; j < 4; ++j) p[j] = (_Float16)kx[i][j];
        *(v4h*)&kT[0][(r0 + i) * KSTR + c0] = p;
    }
#pragma unroll
    for (int j = 0; j < 4; ++j) {
        v4h p;
#pragma unroll
        for (int i = 0; i < 4; ++i) p[i] = (_Float16)vx[i][j];
        *(v4h*)&vT[0][(c0 + j) * KSTR + r0] = p;
    }
    __syncthreads();

    for (int kv = 0; kv < NKV; ++kv) {
        const int p = kv & 1;

        // ---- issue next tile's global loads early (latency hidden by WMMAs)
        if (kv + 1 < NKV) {
            const float* kg = kb + (size_t)((kv + 1) * KVTILE + r0) * HDIM + c0;
            const float* vg = vb + (size_t)((kv + 1) * KVTILE + r0) * HDIM + c0;
#pragma unroll
            for (int i = 0; i < 4; ++i) {
                const float4 a = *(const float4*)(kg + i * HDIM);
                const float4 b = *(const float4*)(vg + i * HDIM);
                kx[i][0] = a.x; kx[i][1] = a.y; kx[i][2] = a.z; kx[i][3] = a.w;
                vx[i][0] = b.x; vx[i][1] = b.y; vx[i][2] = b.z; vx[i][3] = b.w;
            }
            if (kv + 2 < NKV) {                 // warm L2 for tile kv+2
                __builtin_prefetch(kg + KVTILE * HDIM, 0, 3);
                __builtin_prefetch(vg + KVTILE * HDIM, 0, 3);
            }
        }

        const _Float16* kTp = kT[p];
        const _Float16* vTp = vT[p];

        // ---- S^T = K * Q^T : frag j holds keys j*16+{v|v+8}, cols = queries
        v8f s[4];
#pragma unroll
        for (int j = 0; j < 4; ++j)
#pragma unroll
            for (int v = 0; v < 8; ++v) s[j][v] = 0.0f;
#pragma unroll
        for (int kf = 0; kf < 2; ++kf) {
#pragma unroll
            for (int j = 0; j < 4; ++j) {
                const _Float16* base = &kTp[(j * 16 + l16) * KSTR + kf * 32 + half * 16];
                v8h a0 = *(const v8h*)base;
                v8h a1 = *(const v8h*)(base + 8);
                s[j] = __builtin_amdgcn_wmma_f32_16x16x32_f16(
                        false, cat8(a0, a1), false, qb[kf], (short)0, s[j], false, false);
            }
        }

        // ---- online softmax over keys: per-lane scalar stats, tree reductions
        float mj[4];
#pragma unroll
        for (int j = 0; j < 4; ++j) {
            const float a = fmaxf(s[j][0], s[j][1]);
            const float b = fmaxf(s[j][2], s[j][3]);
            const float c = fmaxf(s[j][4], s[j][5]);
            const float d = fmaxf(s[j][6], s[j][7]);
            mj[j] = fmaxf(fmaxf(a, b), fmaxf(c, d));
        }
        float tmax = fmaxf(fmaxf(mj[0], mj[1]), fmaxf(mj[2], mj[3]));
        tmax = fmaxf(tmax, swz16f(tmax));
        const float mn    = fmaxf(mrun, tmax);
        const float alpha = __expf(mrun - mn);
        mrun = mn;
#pragma unroll
        for (int j = 0; j < 4; ++j)
#pragma unroll
            for (int v = 0; v < 8; ++v) s[j][v] = __expf(s[j][v] - mn);
        float sj[4];
#pragma unroll
        for (int j = 0; j < 4; ++j) {
            const float a = s[j][0] + s[j][1];
            const float b = s[j][2] + s[j][3];
            const float c = s[j][4] + s[j][5];
            const float d = s[j][6] + s[j][7];
            sj[j] = (a + b) + (c + d);
        }
        float tsum = (sj[0] + sj[1]) + (sj[2] + sj[3]);
        tsum += swz16f(tsum);
        lrun = fmaf(lrun, alpha, tsum);
#pragma unroll
        for (int t = 0; t < 4; ++t)
#pragma unroll
            for (int v = 0; v < 8; ++v) oacc[t][v] *= alpha;

        // ---- O^T += V^T * P^T ; P^T B-frags built in registers (one swizzle / pair)
#pragma unroll
        for (int kf2 = 0; kf2 < 2; ++kf2) {
            v16h pb;
#pragma unroll
            for (int v = 0; v < 8; ++v) {
                const _Float16 hx = (_Float16)s[2 * kf2][v];
                const _Float16 hy = (_Float16)s[2 * kf2 + 1][v];
                const unsigned int packed =
                    (unsigned int)__builtin_bit_cast(unsigned short, hx) |
                    ((unsigned int)__builtin_bit_cast(unsigned short, hy) << 16);
                const unsigned int pp =
                    (unsigned int)__builtin_amdgcn_ds_swizzle((int)packed, 0x401f);
                const _Float16 plo = __builtin_bit_cast(_Float16, (unsigned short)(pp & 0xffffu));
                const _Float16 phi = __builtin_bit_cast(_Float16, (unsigned short)(pp >> 16));
                pb[v]     = half ? phi : hx;
                pb[8 + v] = half ? hy  : plo;
            }
#pragma unroll
            for (int t = 0; t < 4; ++t) {
                const _Float16* vbase = &vTp[(t * 16 + l16) * KSTR + kf2 * 32 + half * 16];
                v8h a0 = *(const v8h*)vbase;
                v8h a1 = *(const v8h*)(vbase + 8);
                oacc[t] = __builtin_amdgcn_wmma_f32_16x16x32_f16(
                        false, cat8(a0, a1), false, pb, (short)0, oacc[t], false, false);
            }
        }

        // ---- stage next tile into the other buffer (load-wait lands here)
        if (kv + 1 < NKV) {
            _Float16* kTn = kT[p ^ 1];
            _Float16* vTn = vT[p ^ 1];
#pragma unroll
            for (int i = 0; i < 4; ++i) {
                v4h q;
#pragma unroll
                for (int j = 0; j < 4; ++j) q[j] = (_Float16)kx[i][j];
                *(v4h*)&kTn[(r0 + i) * KSTR + c0] = q;
            }
#pragma unroll
            for (int j = 0; j < 4; ++j) {
                v4h q;
#pragma unroll
                for (int i = 0; i < 4; ++i) q[i] = (_Float16)vx[i][j];
                *(v4h*)&vTn[(c0 + j) * KSTR + r0] = q;
            }
        }
        __syncthreads();   // one barrier per iteration (double buffer)
    }

    // ---- epilogue: O^T frag -> each lane owns 8 consecutive d per tile (coalesced)
    const float inv = 1.0f / lrun;
    float* orow = O + ((size_t)bh * SEQ + qbase + wave * 16 + l16) * HDIM;
#pragma unroll
    for (int t = 0; t < 4; ++t)
#pragma unroll
        for (int v = 0; v < 8; ++v)
            orow[t * 16 + half * 8 + v] = oacc[t][v] * inv;
}

extern "C" void kernel_launch(void* const* d_in, const int* in_sizes, int n_in,
                              void* d_out, int out_size, void* d_ws, size_t ws_size,
                              hipStream_t stream) {
    (void)in_sizes; (void)n_in; (void)out_size; (void)d_ws; (void)ws_size;
    const float* Q = (const float*)d_in[0];
    const float* K = (const float*)d_in[1];
    const float* V = (const float*)d_in[2];
    float* O = (float*)d_out;
    dim3 grid(128 /* B*H */, SEQ / QTILE);
    dim3 block(256);
    ScaledDotProductAttention_80659485819079_kernel<<<grid, block, 0, stream>>>(Q, K, V, O);
}
